// AttentionFlow_9758165696816
// MI455X (gfx1250) — matrix operand order repaired and checked by reading, not compiled, not run on hardware
//
#include <hip/hip_runtime.h>
#include <hip/hip_bf16.h>

typedef __attribute__((ext_vector_type(16))) __bf16 v16bf;
typedef __attribute__((ext_vector_type(8)))  __bf16 v8bf;
typedef __attribute__((ext_vector_type(8)))  float  v8f;

#define T_DIM 8192
#define J_DIM 1024
#define DD    512

// round-to-nearest-even f32 -> bf16
__device__ __forceinline__ unsigned short f2bf(float f) {
  unsigned u = __builtin_bit_cast(unsigned, f);
  unsigned r = u + 0x7FFFu + ((u >> 16) & 1u);
  return (unsigned short)(r >> 16);
}

__device__ __forceinline__ v16bf catbf(v8bf lo, v8bf hi) {
  return __builtin_shufflevector(lo, hi, 0,1,2,3,4,5,6,7,8,9,10,11,12,13,14,15);
}

// ---------------- prep 1: row dots: cvec = ctx.wc + Wb ; qvec = qry.wq -------
__global__ __launch_bounds__(128) void k_rowdot(const float* __restrict__ ctx,
                                                const float* __restrict__ qry,
                                                const float* __restrict__ Ww,
                                                const float* __restrict__ Wb,
                                                float* __restrict__ cvec,
                                                float* __restrict__ qvec) {
  int row  = blockIdx.x * 4 + (threadIdx.x >> 5);   // 9216 rows total
  int lane = threadIdx.x & 31;
  const float* src; const float* w;
  if (row < T_DIM) { src = ctx + (size_t)row * DD;           w = Ww; }
  else             { src = qry + (size_t)(row - T_DIM) * DD; w = Ww + DD; }
  float s = 0.f;
  for (int d = lane; d < DD; d += 32) s += src[d] * w[d];
  for (int m = 16; m; m >>= 1) s += __shfl_xor(s, m, 32);
  if (lane == 0) {
    if (row < T_DIM) cvec[row] = s + Wb[0];
    else             qvec[row - T_DIM] = s;
  }
}

// ---------------- prep 2: bf16 casts + transpose + zero q2c ------------------
__global__ __launch_bounds__(256) void k_prep(const float* __restrict__ ctx,
                                              const float* __restrict__ qry,
                                              const float* __restrict__ Ww,
                                              unsigned short* __restrict__ Mbf,
                                              unsigned short* __restrict__ Qbf,
                                              unsigned short* __restrict__ QbfT,
                                              float* __restrict__ q2c) {
  int i = blockIdx.x * 256 + threadIdx.x;           // over T*DD = 4M
  const float* wm = Ww + 2 * DD;
  if (i < T_DIM * DD) {
    int d = i & (DD - 1);
    Mbf[i] = f2bf(ctx[i] * wm[d]);
  }
  if (i < J_DIM * DD) {
    int j = i >> 9, d = i & (DD - 1);
    unsigned short q = f2bf(qry[i]);
    Qbf[i] = q;
    QbfT[(size_t)d * J_DIM + j] = q;
  }
  if (i < DD) q2c[i] = 0.f;
}

// -------- fused flash attention: 16-row tile / WG, 4 waves split J & D -------
__global__ __launch_bounds__(128) void k_flash(const unsigned short* __restrict__ Mbf,
                                               const unsigned short* __restrict__ Qbf,
                                               const unsigned short* __restrict__ QbfT,
                                               const float* __restrict__ cvec,
                                               const float* __restrict__ qvec,
                                               const float* __restrict__ ctx,
                                               float* __restrict__ c2q,
                                               float* __restrict__ q2c) {
  extern __shared__ char smem[];
  float*  S    = (float*)smem;                       // 16 x 1024 f32  (64 KB)
  __bf16* P    = (__bf16*)(smem + 65536);            // 16 x 1024 bf16 (32 KB)
  float*  mrow = (float*)(smem + 65536 + 32768);     // 16
  float*  rl   = mrow + 16;                          // 16
  float*  red1 = rl + 16;                            // 128 (per-seg max)
  float*  red2 = red1 + 128;                         // 128 (per-seg expsum)

  const int tid     = threadIdx.x;
  const int wv      = tid >> 5;                      // wave 0..3
  const int lane    = tid & 31;
  const int lm      = lane & 15;                     // N / M-row within half
  const int h       = lane >> 4;                     // lane half
  const int kh      = h * 8;                         // A-frag K-half offset
  const int rowbase = blockIdx.x * 16;

  // Preload A fragments (same 16 rows for all 4 waves; register-resident)
  const unsigned short* arow = Mbf + (size_t)(rowbase + lm) * DD;
  v16bf A[16];
#pragma unroll
  for (int kf = 0; kf < 16; ++kf) {
    int kb = kf * 32;
    v8bf lo = *(const v8bf*)(arow + kb + kh);
    v8bf hi = *(const v8bf*)(arow + kb + 16 + kh);
    A[kf] = catbf(lo, hi);
  }
  float cvr[8];
#pragma unroll
  for (int r = 0; r < 8; ++r) cvr[r] = cvec[rowbase + r + 8 * h];

  // ---- Pass 1: each wave computes a J-quarter of S -------------------------
  for (int jt = 0; jt < 16; ++jt) {
    int jbase = (wv * 16 + jt) * 16;
    const unsigned short* qrow = Qbf + (size_t)(jbase + lm) * DD;  // B: N=lm
    if (jt < 15)                                                   // L2 prefetch
      __builtin_prefetch(qrow + (size_t)16 * DD, 0, 1);
    v8f acc = {};
#pragma unroll
    for (int kf = 0; kf < 16; ++kf) {
      int kb = kf * 32 + kh;                      // B: table order, K-half offset
      v8bf lo = *(const v8bf*)(qrow + kb);
      v8bf hi = *(const v8bf*)(qrow + kb + 16);
      v16bf B = catbf(lo, hi);
      acc = __builtin_amdgcn_wmma_f32_16x16x32_bf16(false, A[kf], false, B,
                                                    (short)0, acc, false, false);
    }
    float qb = qvec[jbase + lm];
#pragma unroll
    for (int r = 0; r < 8; ++r)
      S[(r + 8 * h) * J_DIM + jbase + lm] = acc[r] + cvr[r] + qb;
  }
  __syncthreads();

  // ---- Pass 1.5: rowmax + exp + unnormalized P (bf16), 8 segs/row ---------
  {
    const int row = tid & 15;
    const int seg = tid >> 4;                      // 0..7, 128 j's each
    const float* srow = S + row * J_DIM + seg * 128;
    float m = -3.4e38f;
    for (int j = 0; j < 128; j += 4) {
      float4 v = *(const float4*)(srow + j);
      m = fmaxf(m, fmaxf(fmaxf(v.x, v.y), fmaxf(v.z, v.w)));
    }
    red1[seg * 16 + row] = m;
    __syncthreads();
    float mm = red1[row];
#pragma unroll
    for (int s2 = 1; s2 < 8; ++s2) mm = fmaxf(mm, red1[s2 * 16 + row]);

    float ls = 0.f;
    __bf16* prow = P + row * J_DIM + seg * 128;
    for (int j = 0; j < 128; j += 4) {
      float4 v = *(const float4*)(srow + j);
      float e0 = __expf(v.x - mm), e1 = __expf(v.y - mm);
      float e2 = __expf(v.z - mm), e3 = __expf(v.w - mm);
      ls += (e0 + e1) + (e2 + e3);
      unsigned long long pk = (unsigned long long)f2bf(e0)
                            | ((unsigned long long)f2bf(e1) << 16)
                            | ((unsigned long long)f2bf(e2) << 32)
                            | ((unsigned long long)f2bf(e3) << 48);
      *(unsigned long long*)(prow + j) = pk;
    }
    red2[seg * 16 + row] = ls;
    __syncthreads();
    if (tid < 16) {
      float l = red2[tid];
#pragma unroll
      for (int s2 = 1; s2 < 8; ++s2) l += red2[s2 * 16 + tid];
      mrow[tid] = mm;                              // tid<16 => row==tid
      rl[tid]   = 1.f / l;
    }
  }
  __syncthreads();

  // ---- q2c contribution: sum_r mrow[r] * ctx[row, d]  (global f32 atomics) -
  for (int d = tid; d < DD; d += 128) {
    float s = 0.f;
#pragma unroll
    for (int r = 0; r < 16; ++r)
      s += mrow[r] * ctx[(size_t)(rowbase + r) * DD + d];
    atomicAdd(q2c + d, s);
  }

  // ---- Pass 2: c2q = (P @ query) * (1/l); 8 d-tiles per wave ---------------
  const __bf16* prow = P + lm * J_DIM;             // A: lane holds row lm
  for (int dt2 = 0; dt2 < 8; ++dt2) {
    int dbase = (wv * 8 + dt2) * 16;
    const unsigned short* brow = QbfT + (size_t)(dbase + lm) * J_DIM; // B: N=lm
    if (dt2 < 7)                                                      // prefetch
      __builtin_prefetch(brow + (size_t)16 * J_DIM, 0, 1);
    v8f acc = {};
#pragma unroll
    for (int jt = 0; jt < 32; ++jt) {
      int jb = jt * 32;
      v8bf alo = *(const v8bf*)(prow + jb + kh);
      v8bf ahi = *(const v8bf*)(prow + jb + 16 + kh);
      v16bf Af = catbf(alo, ahi);
      int kb = jb + kh;
      v8bf blo = *(const v8bf*)(brow + kb);
      v8bf bhi = *(const v8bf*)(brow + kb + 16);
      v16bf Bf = catbf(blo, bhi);
      acc = __builtin_amdgcn_wmma_f32_16x16x32_bf16(false, Af, false, Bf,
                                                    (short)0, acc, false, false);
    }
#pragma unroll
    for (int r = 0; r < 8; ++r) {
      int row = r + 8 * h;
      c2q[(size_t)(rowbase + row) * DD + dbase + lm] = acc[r] * rl[row];
    }
  }
}

// ---------------- finalize: G = [ctx, c2q, ctx*c2q, ctx*q2c] -----------------
__global__ __launch_bounds__(256) void k_final(const float* __restrict__ ctx,
                                               const float* __restrict__ c2q,
                                               const float* __restrict__ q2c,
                                               float* __restrict__ G) {
  int i  = blockIdx.x * 256 + threadIdx.x;        // over T * 128 float4s
  int t  = i >> 7;
  int d4 = (i & 127) << 2;
  float4 c = *(const float4*)(ctx + (size_t)t * DD + d4);
  float4 a = *(const float4*)(c2q + (size_t)t * DD + d4);
  float4 q = *(const float4*)(q2c + d4);
  float* g = G + (size_t)t * (4 * DD);
  *(float4*)(g + d4)            = c;
  *(float4*)(g + DD + d4)       = a;
  *(float4*)(g + 2 * DD + d4)   = make_float4(c.x * a.x, c.y * a.y, c.z * a.z, c.w * a.w);
  *(float4*)(g + 3 * DD + d4)   = make_float4(c.x * q.x, c.y * q.y, c.z * q.z, c.w * q.w);
}

extern "C" void kernel_launch(void* const* d_in, const int* in_sizes, int n_in,
                              void* d_out, int out_size, void* d_ws, size_t ws_size,
                              hipStream_t stream) {
  (void)in_sizes; (void)n_in; (void)out_size; (void)ws_size;
  const float* ctx = (const float*)d_in[0];
  const float* qry = (const float*)d_in[1];
  const float* Ww  = (const float*)d_in[2];
  const float* Wb  = (const float*)d_in[3];
  float* G = (float*)d_out;

  char* ws = (char*)d_ws;
  unsigned short* Mbf  = (unsigned short*)(ws);              //  8 MB
  unsigned short* Qbf  = (unsigned short*)(ws +  8388608);   //  1 MB
  unsigned short* QbfT = (unsigned short*)(ws +  9437184);   //  1 MB
  float* cvec = (float*)(ws + 10485760);                     // 32 KB
  float* qvec = (float*)(ws + 10518528);                     //  4 KB
  float* q2c  = (float*)(ws + 10522624);                     //  2 KB
  float* c2q  = (float*)(ws + 10524672);                     // 16 MB

  k_rowdot<<<(T_DIM + J_DIM) / 4, 128, 0, stream>>>(ctx, qry, Ww, Wb, cvec, qvec);
  k_prep<<<(T_DIM * DD) / 256, 256, 0, stream>>>(ctx, qry, Ww, Mbf, Qbf, QbfT, q2c);

  const int SMEM = 65536 + 32768 + 2 * 16 * 4 + 2 * 128 * 4; // ~99.4 KB / WG
  hipFuncSetAttribute(reinterpret_cast<const void*>(&k_flash),
                      hipFuncAttributeMaxDynamicSharedMemorySize, SMEM);
  k_flash<<<T_DIM / 16, 128, SMEM, stream>>>(Mbf, Qbf, QbfT, cvec, qvec, ctx, c2q, q2c);

  k_final<<<(T_DIM * 128) / 256, 256, 0, stream>>>(ctx, c2q, q2c, G);
}
